// EquiBlock3Body_50929722196748
// MI455X (gfx1250) — matrix operand order repaired and checked by reading, not compiled, not run on hardware
//
#include <hip/hip_runtime.h>
#include <hip/hip_bf16.h>

typedef __attribute__((ext_vector_type(16))) _Float16 v16h;
typedef __attribute__((ext_vector_type(8)))  _Float16 v8h;
typedef __attribute__((ext_vector_type(4)))  _Float16 v4h;
typedef __attribute__((ext_vector_type(8)))  float    v8f;

#define HDIM  64
#define NW    4           // waves per block
#define BLOCK (NW * 32)

__device__ __forceinline__ float silu_f(float x) { return x / (1.f + __expf(-x)); }
__device__ __forceinline__ float sigm_f(float x) { return 1.f / (1.f + __expf(-x)); }

// ---- WMMA operand helpers (CDNA5 wave32, v_wmma_f32_16x16x32_f16) ----
// A 16x32 f16: M = lane%16 (both halves). Per lane the 16 elements are two
// contiguous 8-element K runs: [k0 + half*8 .. +7] and [k0 + 16 + half*8 .. +7].
__device__ __forceinline__ v16h ldac(const _Float16* row, int half) {
  v8h lo = *(const v8h*)(row + half * 8);
  v8h hi = *(const v8h*)(row + 16 + half * 8);
  v16h a;
#pragma unroll
  for (int e = 0; e < 8; ++e) { a[e] = lo[e]; a[e + 8] = hi[e]; }
  return a;
}
// B operands are pre-swizzled by k_prep into [ntile][kchunk][lane][16]: one
// contiguous 32B load per lane.
__device__ __forceinline__ v16h ldbp(const _Float16* Bp, int Kc, int nt, int kc, int lane) {
  return *((const v16h*)Bp + ((size_t)(nt * Kc + kc) * 32 + lane));
}
// C/D f32 16x16: element r of v8f is (m = r + 8*half, n = lane%16)

__device__ __forceinline__ void stage_row4(_Float16* sth, int m, int q, float4 val) {
  v4h h;
  h[0] = (_Float16)val.x; h[1] = (_Float16)val.y;
  h[2] = (_Float16)val.z; h[3] = (_Float16)val.w;
  *(v4h*)&sth[m * 64 + q * 4] = h;
}

// ---- weight conversion: f32 row-major -> f16 pre-swizzled WMMA B layout ----
// packed index: (((nt*Kc + kc)*32 + lane)*16 + e) <-> W[kc*32 + (e>>3)*16 + (lane>>4)*8 + (e&7)][nt*16 + (lane&15)]
__global__ void k_prep(const float* __restrict__ Wt1, const float* __restrict__ Wt2,
                       const float* __restrict__ Wm1, const float* __restrict__ Wm2,
                       const float* __restrict__ Wu1, const float* __restrict__ Wu2,
                       const float* __restrict__ Wg1, const float* __restrict__ Wg2,
                       _Float16* __restrict__ dst) {
  const float* srcs[8] = {Wt1, Wt2, Wm1, Wm2, Wu1, Wu2, Wg1, Wg2};
  const int ks[8] = {144, 64, 256, 64, 192, 64, 192, 64};
  const int kp[8] = {160, 64, 256, 64, 192, 64, 192, 64};
  const int nn[8] = {64, 64, 64, 128, 64, 64, 64, 64};
  int i = blockIdx.x * blockDim.x + threadIdx.x;
  int off = 0;
  for (int r = 0; r < 8; ++r) {
    int cnt = kp[r] * nn[r];
    if (i < off + cnt) {
      int li = i - off;
      int e = li & 15;
      int lane = (li >> 4) & 31;
      int chunk = li >> 9;
      int Kc = kp[r] >> 5;
      int kc = chunk % Kc, nt = chunk / Kc;
      int half = lane >> 4;
      int n = nt * 16 + (lane & 15);
      int k = kc * 32 + (e >> 3) * 16 + half * 8 + (e & 7);
      float val = (k < ks[r]) ? srcs[r][k * nn[r] + n] : 0.f;
      dst[i] = (_Float16)val;
      return;
    }
    off += cnt;
  }
}

// ---- kernel 1: triplet MLP + scatter into angle_agg[E,64] ----
__global__ __launch_bounds__(BLOCK) void k_triplet(
    int Tn, const float* __restrict__ rbf, const int* __restrict__ kj,
    const int* __restrict__ ji, const float* __restrict__ ang,
    const _Float16* __restrict__ Wt1, const float* __restrict__ bt1,
    const _Float16* __restrict__ Wt2, const float* __restrict__ bt2,
    float* __restrict__ angle_agg) {
  __shared__ __align__(16) _Float16 stg [NW][16 * 64];
  __shared__ __align__(16) _Float16 hbuf[NW][16 * 64];
  const int lane = threadIdx.x & 31, wave = threadIdx.x >> 5;
  const int half = lane >> 4, m16 = lane & 15, nloc = lane & 15;
  const int t0 = (blockIdx.x * NW + wave) * 16;
  _Float16* st = stg[wave];
  _Float16* hb = hbuf[wave];

  v16h A[5];
  // section 0: rbf[kj[t]] (cols 0..63)
#pragma unroll
  for (int j = 0; j < 8; ++j) {
    int slot = lane + 32 * j, m = slot >> 4, q = slot & 15;
    int t = t0 + m; if (t > Tn - 1) t = Tn - 1;
    const float* rp = rbf + (size_t)kj[t] * HDIM;
    stage_row4(st, m, q, *(const float4*)(rp + q * 4));
  }
  __syncthreads();
  A[0] = ldac(&st[m16 * 64 + 0], half);
  A[1] = ldac(&st[m16 * 64 + 32], half);
  __syncthreads();
  // section 1: rbf[ji[t]] (cols 64..127)
#pragma unroll
  for (int j = 0; j < 8; ++j) {
    int slot = lane + 32 * j, m = slot >> 4, q = slot & 15;
    int t = t0 + m; if (t > Tn - 1) t = Tn - 1;
    const float* rp = rbf + (size_t)ji[t] * HDIM;
    stage_row4(st, m, q, *(const float4*)(rp + q * 4));
  }
  __syncthreads();
  A[2] = ldac(&st[m16 * 64 + 0], half);
  A[3] = ldac(&st[m16 * 64 + 32], half);
  __syncthreads();
  // section 2: angle(16) + zero pad(16) (cols 128..159)
#pragma unroll
  for (int j = 0; j < 4; ++j) {
    int slot = lane + 32 * j, m = slot >> 3, q = slot & 7;
    int t = t0 + m; if (t > Tn - 1) t = Tn - 1;
    float4 val = make_float4(0.f, 0.f, 0.f, 0.f);
    if (q < 4) val = *(const float4*)(ang + (size_t)t * 16 + q * 4);
    stage_row4(st, m, q, val);
  }
  __syncthreads();
  A[4] = ldac(&st[m16 * 64 + 0], half);

  // layer 1: [16,160] x [160,64]  (Kc=5, 4 n-tiles)
#pragma unroll
  for (int nt = 0; nt < 4; ++nt) {
    v8f c = {};
#pragma unroll
    for (int kc = 0; kc < 5; ++kc) {
      v16h b = ldbp(Wt1, 5, nt, kc, lane);
      c = __builtin_amdgcn_wmma_f32_16x16x32_f16(false, A[kc], false, b, (short)0, c, false, false);
    }
    int n = nt * 16 + nloc;
    float bias = bt1[n];
#pragma unroll
    for (int r = 0; r < 8; ++r)
      hb[(r + 8 * half) * 64 + n] = (_Float16)silu_f(c[r] + bias);
  }
  __syncthreads();
  v16h A2[2];
  A2[0] = ldac(&hb[m16 * 64 + 0], half);
  A2[1] = ldac(&hb[m16 * 64 + 32], half);

  int jir[8]; bool ok8[8];
#pragma unroll
  for (int r = 0; r < 8; ++r) {
    int t = t0 + r + 8 * half;
    ok8[r] = (t < Tn);
    jir[r] = ji[t < Tn ? t : Tn - 1];
  }
  // layer 2 + scatter (segment_sum over ji)
#pragma unroll
  for (int nt = 0; nt < 4; ++nt) {
    v8f c = {};
#pragma unroll
    for (int kc = 0; kc < 2; ++kc) {
      v16h b = ldbp(Wt2, 2, nt, kc, lane);
      c = __builtin_amdgcn_wmma_f32_16x16x32_f16(false, A2[kc], false, b, (short)0, c, false, false);
    }
    int n = nt * 16 + nloc;
    float bias = bt2[n];
#pragma unroll
    for (int r = 0; r < 8; ++r)
      if (ok8[r]) atomicAdd(&angle_agg[(size_t)jir[r] * HDIM + n], c[r] + bias);
  }
}

// ---- kernel 2: edge message MLP + scatter into agg_s / agg_v ----
__global__ __launch_bounds__(BLOCK) void k_edge(
    int En, const float* __restrict__ s, const int* __restrict__ eidx,
    const float* __restrict__ rbf, const float* __restrict__ dir,
    const float* __restrict__ angle_agg,
    const _Float16* __restrict__ Wm1, const float* __restrict__ bm1,
    const _Float16* __restrict__ Wm2, const float* __restrict__ bm2,
    float* __restrict__ agg_s, float* __restrict__ agg_v) {
  __shared__ __align__(16) _Float16 stg [NW][16 * 64];
  __shared__ __align__(16) _Float16 hbuf[NW][16 * 64];
  const int lane = threadIdx.x & 31, wave = threadIdx.x >> 5;
  const int half = lane >> 4, m16 = lane & 15, nloc = lane & 15;
  const int e0 = (blockIdx.x * NW + wave) * 16;
  _Float16* st = stg[wave];
  _Float16* hb = hbuf[wave];

  v16h A[8];
#pragma unroll
  for (int sec = 0; sec < 4; ++sec) {
#pragma unroll
    for (int j = 0; j < 8; ++j) {
      int slot = lane + 32 * j, m = slot >> 4, q = slot & 15;
      int e = e0 + m; if (e > En - 1) e = En - 1;
      const float* rp;
      if (sec == 0)      rp = s + (size_t)eidx[e] * HDIM;        // s[src]
      else if (sec == 1) rp = s + (size_t)eidx[En + e] * HDIM;   // s[dst]
      else if (sec == 2) rp = rbf + (size_t)e * HDIM;            // edge_rbf
      else               rp = angle_agg + (size_t)e * HDIM;      // angle_agg
      stage_row4(st, m, q, *(const float4*)(rp + q * 4));
    }
    __syncthreads();
    A[2 * sec]     = ldac(&st[m16 * 64 + 0], half);
    A[2 * sec + 1] = ldac(&st[m16 * 64 + 32], half);
    __syncthreads();
  }

  // layer 1: [16,256] x [256,64]  (Kc=8)
#pragma unroll
  for (int nt = 0; nt < 4; ++nt) {
    v8f c = {};
#pragma unroll
    for (int kc = 0; kc < 8; ++kc) {
      v16h b = ldbp(Wm1, 8, nt, kc, lane);
      c = __builtin_amdgcn_wmma_f32_16x16x32_f16(false, A[kc], false, b, (short)0, c, false, false);
    }
    int n = nt * 16 + nloc;
    float bias = bm1[n];
#pragma unroll
    for (int r = 0; r < 8; ++r)
      hb[(r + 8 * half) * 64 + n] = (_Float16)silu_f(c[r] + bias);
  }
  __syncthreads();
  v16h A2[2];
  A2[0] = ldac(&hb[m16 * 64 + 0], half);
  A2[1] = ldac(&hb[m16 * 64 + 32], half);

  int dstm[8]; bool ok8[8]; float dx[8], dy[8], dz[8];
#pragma unroll
  for (int r = 0; r < 8; ++r) {
    int e = e0 + r + 8 * half;
    ok8[r] = (e < En);
    int ec = e < En ? e : En - 1;
    dstm[r] = eidx[En + ec];
    dx[r] = dir[(size_t)ec * 3 + 0];
    dy[r] = dir[(size_t)ec * 3 + 1];
    dz[r] = dir[(size_t)ec * 3 + 2];
  }
  // layer 2: [16,64] x [64,128]; cols 0..63 -> m_s, 64..127 -> m_v coeff
#pragma unroll
  for (int nt = 0; nt < 8; ++nt) {
    v8f c = {};
#pragma unroll
    for (int kc = 0; kc < 2; ++kc) {
      v16h b = ldbp(Wm2, 2, nt, kc, lane);
      c = __builtin_amdgcn_wmma_f32_16x16x32_f16(false, A2[kc], false, b, (short)0, c, false, false);
    }
    int n = nt * 16 + nloc;
    float bias = bm2[n];
    if (nt < 4) {
#pragma unroll
      for (int r = 0; r < 8; ++r)
        if (ok8[r]) atomicAdd(&agg_s[(size_t)dstm[r] * HDIM + n], c[r] + bias);
    } else {
      int nh = n - 64;
#pragma unroll
      for (int r = 0; r < 8; ++r)
        if (ok8[r]) {
          float coef = c[r] + bias;
          size_t base = (size_t)dstm[r] * (HDIM * 3) + nh * 3;
          atomicAdd(&agg_v[base + 0], coef * dx[r]);
          atomicAdd(&agg_v[base + 1], coef * dy[r]);
          atomicAdd(&agg_v[base + 2], coef * dz[r]);
        }
    }
  }
}

// ---- kernel 3: node update MLPs + gate + LayerNorm + silu ----
__global__ __launch_bounds__(BLOCK) void k_node(
    int Nn, const float* __restrict__ s, const float* __restrict__ v,
    const float* __restrict__ agg_s, const float* __restrict__ agg_v,
    const _Float16* __restrict__ Wu1, const float* __restrict__ bu1,
    const _Float16* __restrict__ Wu2, const float* __restrict__ bu2,
    const _Float16* __restrict__ Wg1, const float* __restrict__ bg1,
    const _Float16* __restrict__ Wg2, const float* __restrict__ bg2,
    const float* __restrict__ lng, const float* __restrict__ lnb,
    float* __restrict__ out_s, float* __restrict__ out_v) {
  __shared__ __align__(16) _Float16 stg  [NW][16 * 64];
  __shared__ __align__(16) _Float16 hbuf [NW][16 * 64];
  __shared__ __align__(16) float    snbuf[NW][16 * 64];
  __shared__ float mubuf[NW][16], rsbuf[NW][16];
  const int lane = threadIdx.x & 31, wave = threadIdx.x >> 5;
  const int half = lane >> 4, m16 = lane & 15, nloc = lane & 15;
  const int n0 = (blockIdx.x * NW + wave) * 16;
  _Float16* st = stg[wave];
  _Float16* hb = hbuf[wave];
  float* snb = snbuf[wave];

  v16h A[6];
#pragma unroll
  for (int sec = 0; sec < 2; ++sec) {
#pragma unroll
    for (int j = 0; j < 8; ++j) {
      int slot = lane + 32 * j, m = slot >> 4, q = slot & 15;
      int node = n0 + m; if (node > Nn - 1) node = Nn - 1;
      const float* rp = (sec == 0 ? s : agg_s) + (size_t)node * HDIM;
      stage_row4(st, m, q, *(const float4*)(rp + q * 4));
    }
    __syncthreads();
    A[2 * sec]     = ldac(&st[m16 * 64 + 0], half);
    A[2 * sec + 1] = ldac(&st[m16 * 64 + 32], half);
    __syncthreads();
  }
  // section 2: v_norm
#pragma unroll
  for (int j = 0; j < 32; ++j) {
    int slot = lane + 32 * j, m = slot >> 6, h = slot & 63;
    int node = n0 + m; if (node > Nn - 1) node = Nn - 1;
    const float* vp = v + ((size_t)node * HDIM + h) * 3;
    float x = vp[0], y = vp[1], z = vp[2];
    st[m * 64 + h] = (_Float16)sqrtf(x * x + y * y + z * z);
  }
  __syncthreads();
  A[4] = ldac(&st[m16 * 64 + 0], half);
  A[5] = ldac(&st[m16 * 64 + 32], half);
  __syncthreads();

  // ---- MLP u ---- (Kc=6)
#pragma unroll
  for (int nt = 0; nt < 4; ++nt) {
    v8f c = {};
#pragma unroll
    for (int kc = 0; kc < 6; ++kc) {
      v16h b = ldbp(Wu1, 6, nt, kc, lane);
      c = __builtin_amdgcn_wmma_f32_16x16x32_f16(false, A[kc], false, b, (short)0, c, false, false);
    }
    int n = nt * 16 + nloc;
    float bias = bu1[n];
#pragma unroll
    for (int r = 0; r < 8; ++r)
      hb[(r + 8 * half) * 64 + n] = (_Float16)silu_f(c[r] + bias);
  }
  __syncthreads();
  v16h AU[2];
  AU[0] = ldac(&hb[m16 * 64 + 0], half);
  AU[1] = ldac(&hb[m16 * 64 + 32], half);
  __syncthreads();
  v8f su[4];
#pragma unroll
  for (int nt = 0; nt < 4; ++nt) {
    v8f c = {};
#pragma unroll
    for (int kc = 0; kc < 2; ++kc) {
      v16h b = ldbp(Wu2, 2, nt, kc, lane);
      c = __builtin_amdgcn_wmma_f32_16x16x32_f16(false, AU[kc], false, b, (short)0, c, false, false);
    }
    su[nt] = c;
  }
  // ---- MLP g hidden ----
#pragma unroll
  for (int nt = 0; nt < 4; ++nt) {
    v8f c = {};
#pragma unroll
    for (int kc = 0; kc < 6; ++kc) {
      v16h b = ldbp(Wg1, 6, nt, kc, lane);
      c = __builtin_amdgcn_wmma_f32_16x16x32_f16(false, A[kc], false, b, (short)0, c, false, false);
    }
    int n = nt * 16 + nloc;
    float bias = bg1[n];
#pragma unroll
    for (int r = 0; r < 8; ++r)
      hb[(r + 8 * half) * 64 + n] = (_Float16)silu_f(c[r] + bias);
  }
  __syncthreads();
  v16h AG[2];
  AG[0] = ldac(&hb[m16 * 64 + 0], half);
  AG[1] = ldac(&hb[m16 * 64 + 32], half);

  int noder[8]; bool ok8[8];
#pragma unroll
  for (int r = 0; r < 8; ++r) {
    int node = n0 + r + 8 * half;
    ok8[r] = (node < Nn);
    noder[r] = node < Nn ? node : Nn - 1;
  }
  // gate GEMM + s_new + v_new; stash s_new in f32 LDS for LayerNorm
#pragma unroll
  for (int nt = 0; nt < 4; ++nt) {
    v8f c = {};
#pragma unroll
    for (int kc = 0; kc < 2; ++kc) {
      v16h b = ldbp(Wg2, 2, nt, kc, lane);
      c = __builtin_amdgcn_wmma_f32_16x16x32_f16(false, AG[kc], false, b, (short)0, c, false, false);
    }
    int n = nt * 16 + nloc;
    float bu2v = bu2[n], bg2v = bg2[n];
#pragma unroll
    for (int r = 0; r < 8; ++r) {
      int m = r + 8 * half;
      int node = noder[r];
      float sold = s[(size_t)node * HDIM + n];
      float snew = sold + su[nt][r] + bu2v;
      snb[m * 64 + n] = snew;
      float gate = sigm_f(c[r] + bg2v);
      if (ok8[r]) {
        size_t base = ((size_t)node * HDIM + n) * 3;
        out_v[base + 0] = v[base + 0] + gate * agg_v[base + 0];
        out_v[base + 1] = v[base + 1] + gate * agg_v[base + 1];
        out_v[base + 2] = v[base + 2] + gate * agg_v[base + 2];
      }
    }
  }
  __syncthreads();
  if (lane < 16) {
    float sum = 0.f, ss = 0.f;
    for (int h = 0; h < 64; ++h) { float x = snb[lane * 64 + h]; sum += x; ss += x * x; }
    float mu = sum * (1.f / 64.f);
    float var = ss * (1.f / 64.f) - mu * mu;
    mubuf[wave][lane] = mu;
    rsbuf[wave][lane] = rsqrtf(var + 1e-5f);
  }
  __syncthreads();
#pragma unroll
  for (int nt = 0; nt < 4; ++nt) {
    int n = nt * 16 + nloc;
    float g = lng[n], bb = lnb[n];
#pragma unroll
    for (int r = 0; r < 8; ++r) {
      int m = r + 8 * half;
      if (ok8[r]) {
        float x = snb[m * 64 + n];
        float xn = (x - mubuf[wave][m]) * rsbuf[wave][m] * g + bb;
        out_s[(size_t)noder[r] * HDIM + n] = silu_f(xn);
      }
    }
  }
}

extern "C" void kernel_launch(void* const* d_in, const int* in_sizes, int n_in,
                              void* d_out, int out_size, void* d_ws, size_t ws_size,
                              hipStream_t stream) {
  const float* s   = (const float*)d_in[0];
  const float* v   = (const float*)d_in[1];
  const int*   eix = (const int*)  d_in[2];
  const float* rbf = (const float*)d_in[3];
  const float* dir = (const float*)d_in[4];
  const int*   tkj = (const int*)  d_in[5];
  const int*   tji = (const int*)  d_in[6];
  const float* ang = (const float*)d_in[7];
  const float* Wt1 = (const float*)d_in[8],  *bt1 = (const float*)d_in[9];
  const float* Wt2 = (const float*)d_in[10], *bt2 = (const float*)d_in[11];
  const float* Wm1 = (const float*)d_in[12], *bm1 = (const float*)d_in[13];
  const float* Wm2 = (const float*)d_in[14], *bm2 = (const float*)d_in[15];
  const float* Wu1 = (const float*)d_in[16], *bu1 = (const float*)d_in[17];
  const float* Wu2 = (const float*)d_in[18], *bu2 = (const float*)d_in[19];
  const float* Wg1 = (const float*)d_in[20], *bg1 = (const float*)d_in[21];
  const float* Wg2 = (const float*)d_in[22], *bg2 = (const float*)d_in[23];
  const float* lng = (const float*)d_in[24], *lnb = (const float*)d_in[25];
  (void)n_in; (void)out_size; (void)ws_size;

  const int N = in_sizes[0] / HDIM;
  const int E = in_sizes[3] / HDIM;
  const int T = in_sizes[5];

  // workspace layout: [pre-swizzled f16 weights | angle_agg E*64 | agg_s N*64 | agg_v N*192]
  _Float16* w16 = (_Float16*)d_ws;
  const size_t OFF_T1 = 0;
  const size_t OFF_T2 = OFF_T1 + 160 * 64;
  const size_t OFF_M1 = OFF_T2 + 64 * 64;
  const size_t OFF_M2 = OFF_M1 + 256 * 64;
  const size_t OFF_U1 = OFF_M2 + 64 * 128;
  const size_t OFF_U2 = OFF_U1 + 192 * 64;
  const size_t OFF_G1 = OFF_U2 + 64 * 64;
  const size_t OFF_G2 = OFF_G1 + 192 * 64;
  const size_t WTOT   = OFF_G2 + 64 * 64;   // 71680 f16 elements

  char* p = (char*)d_ws + ((WTOT * sizeof(_Float16) + 255) / 256) * 256;
  float* angle_agg = (float*)p;
  float* aggs = angle_agg + (size_t)E * HDIM;
  float* aggv = aggs + (size_t)N * HDIM;

  hipMemsetAsync(angle_agg, 0,
                 ((size_t)E * HDIM + (size_t)N * HDIM * 4) * sizeof(float), stream);

  k_prep<<<(int)((WTOT + 255) / 256), 256, 0, stream>>>(
      Wt1, Wt2, Wm1, Wm2, Wu1, Wu2, Wg1, Wg2, w16);

  k_triplet<<<(T + 63) / 64, BLOCK, 0, stream>>>(
      T, rbf, tkj, tji, ang, w16 + OFF_T1, bt1, w16 + OFF_T2, bt2, angle_agg);

  k_edge<<<(E + 63) / 64, BLOCK, 0, stream>>>(
      E, s, eix, rbf, dir, angle_agg, w16 + OFF_M1, bm1, w16 + OFF_M2, bm2, aggs, aggv);

  float* out_s = (float*)d_out;
  float* out_v = out_s + (size_t)N * HDIM;
  k_node<<<(N + 63) / 64, BLOCK, 0, stream>>>(
      N, s, v, aggs, aggv, w16 + OFF_U1, bu1, w16 + OFF_U2, bu2,
      w16 + OFF_G1, bg1, w16 + OFF_G2, bg2, lng, lnb, out_s, out_v);
}
